// EnBiMambaLayer_77206332113400
// MI455X (gfx1250) — compile-verified, matrix operand order checked by reading, not run on hardware
//
#include <hip/hip_runtime.h>
#include <hip/hip_bf16.h>
#include <math.h>

// ---------------------------------------------------------------------------
// BiMamba layer for gfx1250 (MI455X). All GEMMs via v_wmma_f32_16x16x32_bf16
// with bf16 operands pre-staged in memory (no conversion VALU in the K-loop),
// 16x64 strip per wave (1 A-frag -> 4 WMMAs per K-step).
// B=4, L=2048, D_MODEL=512, D_INNER=1024, D_STATE=16, DT_RANK=32, D_CONV=4
// ---------------------------------------------------------------------------

typedef __attribute__((ext_vector_type(16))) __bf16 v16bf;
typedef __attribute__((ext_vector_type(8)))  float  v8f;

#define NB   4
#define LSEQ 2048
#define DMOD 512
#define DINR 1024
#define NST  16
#define DTRK 32
#define ROWS (NB * LSEQ)          /* 8192 rows for all GEMMs */

// fp32 -> bf16 bits with round-to-nearest-even
__device__ __forceinline__ unsigned short bf_bits(float f) {
    union { float f; unsigned u; } v; v.f = f;
    unsigned r = v.u + 0x7FFFu + ((v.u >> 16) & 1u);
    return (unsigned short)(r >> 16);
}

__device__ __forceinline__ float silu_f(float x) {
    return x / (1.0f + __expf(-x));
}

// Load one 16x32 bf16 WMMA fragment from row-major bf16 (ushort) memory.
// ISA layout (16-bit A/B 16x32): lane l -> row = l&15, half = l>>4;
// element i -> K = (i<8 ? k0+half*8+i : k0+16+half*8+(i-8)).
// => exactly two contiguous 16-byte loads per lane.
__device__ __forceinline__ v16bf load_frag_bf(const unsigned short* __restrict__ base,
                                              int row0, int ld, int k0) {
    const int lane = threadIdx.x & 31;
    const unsigned short* p =
        base + (size_t)(row0 + (lane & 15)) * ld + k0 + (lane >> 4) * 8;
    uint4 q[2];
    q[0] = *reinterpret_cast<const uint4*>(p);        // elements 0..7
    q[1] = *reinterpret_cast<const uint4*>(p + 16);   // elements 8..15
    v16bf a;
    __builtin_memcpy(&a, q, 32);
    return a;
}

// Generic WMMA GEMM: C = A * W^T, A:[M,K] bf16, W:[N,K] bf16, C fp32.
// One wave computes a 16x64 strip (4 N-tiles share one A fragment).
// mode 0: store fp32. mode 1: +bias, softplus, store fp32.
// mode 2: store fp32 AND bf16 copy (for downstream GEMM consumption).
__global__ void __launch_bounds__(256)
gemm_wmma_kernel(const unsigned short* __restrict__ A, int lda,
                 const unsigned short* __restrict__ W, int ldb,
                 float* __restrict__ C, int ldc,
                 int M, int N, int K, int mode,
                 const float* __restrict__ bias,
                 unsigned short* __restrict__ Cbf) {
    const int sid     = (int)((blockIdx.x * blockDim.x + threadIdx.x) >> 5);
    const int stripsN = N >> 6;                       // strips of 64 along N
    if (sid >= (M >> 4) * stripsN) return;            // wave-uniform guard
    const int m0 = (sid / stripsN) * 16;
    const int n0 = (sid % stripsN) * 64;

    v8f acc[4] = {v8f{}, v8f{}, v8f{}, v8f{}};
    for (int k0 = 0; k0 < K; k0 += 32) {
        const v16bf a  = load_frag_bf(A, m0, lda, k0);
        const v16bf b0 = load_frag_bf(W, n0,      ldb, k0);
        const v16bf b1 = load_frag_bf(W, n0 + 16, ldb, k0);
        const v16bf b2 = load_frag_bf(W, n0 + 32, ldb, k0);
        const v16bf b3 = load_frag_bf(W, n0 + 48, ldb, k0);
        acc[0] = __builtin_amdgcn_wmma_f32_16x16x32_bf16(false, a, false, b0, (short)0, acc[0], false, false);
        acc[1] = __builtin_amdgcn_wmma_f32_16x16x32_bf16(false, a, false, b1, (short)0, acc[1], false, false);
        acc[2] = __builtin_amdgcn_wmma_f32_16x16x32_bf16(false, a, false, b2, (short)0, acc[2], false, false);
        acc[3] = __builtin_amdgcn_wmma_f32_16x16x32_bf16(false, a, false, b3, (short)0, acc[3], false, false);
    }

    // C/D layout: VGPR i of lane l -> M = m0 + (l>>4)*8 + i, N = n0 + (l&15)
    const int lane = threadIdx.x & 31;
    const int mr   = m0 + (lane >> 4) * 8;
#pragma unroll
    for (int j = 0; j < 4; ++j) {
        const int nc = n0 + j * 16 + (lane & 15);
        if (mode == 1) {
            const float bv = bias[nc];
#pragma unroll
            for (int i = 0; i < 8; ++i) {
                float v = acc[j][i] + bv;
                v = (v > 20.0f) ? v : log1pf(__expf(v));       // softplus
                C[(size_t)(mr + i) * ldc + nc] = v;
            }
        } else {
#pragma unroll
            for (int i = 0; i < 8; ++i) {
                const float v = acc[j][i];
                C[(size_t)(mr + i) * ldc + nc] = v;
                if (mode == 2) Cbf[(size_t)(mr + i) * ldc + nc] = bf_bits(v);
            }
        }
    }
}

// fp32 -> bf16 conversion pass (weights / input staging)
__global__ void __launch_bounds__(256)
cvt_bf16_kernel(const float* __restrict__ in, unsigned short* __restrict__ out, int n) {
    const int i = (int)(blockIdx.x * 256 + threadIdx.x);
    if (i < n) out[i] = bf_bits(in[i]);
}

// Depthwise causal conv (k=4) + SiLU. dir=1 consumes time-reversed xin and
// stores in reversed-time order. Emits fp32 (for scan) and bf16 (for x_proj).
__global__ void __launch_bounds__(256)
conv_silu_kernel(const float* __restrict__ xz,      // [ROWS, 2048], xin = cols 0..1023
                 const float* __restrict__ w,       // [DINR, 4]
                 const float* __restrict__ bias,    // [DINR]
                 float* __restrict__ xc32,          // [ROWS, DINR]
                 unsigned short* __restrict__ xcbf, // [ROWS, DINR]
                 int dir) {
    const int idx = (int)(blockIdx.x * 256 + threadIdx.x);   // over ROWS*DINR
    const int d   = idx & (DINR - 1);
    const int row = idx >> 10;
    const int t   = row & (LSEQ - 1);
    const int b   = row >> 11;
    float s = bias[d];
#pragma unroll
    for (int k = 0; k < 4; ++k) {
        const int sidx = t - 3 + k;
        if (sidx >= 0) {
            const int lsrc = dir ? (LSEQ - 1 - sidx) : sidx;
            s += w[d * 4 + k] * xz[((size_t)(b * LSEQ + lsrc)) * 2048 + d];
        }
    }
    const float v = silu_f(s);
    xc32[idx] = v;
    xcbf[idx] = bf_bits(v);
}

// Selective scan. Each half-wave (16 lanes) owns one d-channel; lane = state n.
// Cross-lane sum over n via shfl_xor (wave32). Fused +x*D and SiLU(z) gate.
// dir=0 stores fp32 into u32; dir=1 adds its (time-flipped) contribution and
// emits the final bf16 u for the out_proj WMMA GEMM.
__global__ void __launch_bounds__(256)
scan_kernel(const float* __restrict__ dtb,   // [ROWS, DINR] softplus(dt)
            const float* __restrict__ xcb,   // [ROWS, DINR]
            const float* __restrict__ dblb,  // [ROWS, 64]  (dt|B|C)
            const float* __restrict__ xz,    // [ROWS, 2048] (z = cols 1024..2047)
            const float* __restrict__ A_log, // [DINR, NST]
            const float* __restrict__ Dv,    // [DINR]
            float* __restrict__ u32,         // [ROWS, DINR]
            unsigned short* __restrict__ ubf,// [ROWS, DINR]
            int dir) {
    const int b    = (int)blockIdx.x / (DINR / 16);
    const int dblk = ((int)blockIdx.x % (DINR / 16)) * 16;
    const int wave = threadIdx.x >> 5;
    const int lane = threadIdx.x & 31;
    const int d    = dblk + wave * 2 + (lane >> 4);
    const int n    = lane & 15;
    const float Ad = -__expf(A_log[d * NST + n]);
    const float Dd = Dv[d];
    const size_t rowbase = (size_t)b * LSEQ;
    float h = 0.0f;
    for (int t = 0; t < LSEQ; ++t) {
        const size_t r = rowbase + t;
        const float dt = dtb[r * DINR + d];
        const float xv = xcb[r * DINR + d];
        const float Bv = dblb[r * 64 + DTRK + n];
        const float Cv = dblb[r * 64 + DTRK + NST + n];
        h = __expf(dt * Ad) * h + (dt * xv) * Bv;
        float p = h * Cv;
        p += __shfl_xor(p, 1, 32);
        p += __shfl_xor(p, 2, 32);
        p += __shfl_xor(p, 4, 32);
        p += __shfl_xor(p, 8, 32);               // sum over 16 states
        if (n == 0) {
            const int to = dir ? (LSEQ - 1 - t) : t;   // flip back for backward dir
            const size_t ro = rowbase + to;
            const float zv = xz[ro * 2048 + DINR + d]; // z at original time index
            const float val = silu_f(zv) * (p + xv * Dd);
            if (dir) {
                const float tot = u32[ro * DINR + d] + val;
                ubf[ro * DINR + d] = bf_bits(tot);
            } else {
                u32[ro * DINR + d] = val;
            }
        }
    }
}

// LayerNorm(y) * g + b, plus residual x. One block per 512-wide row.
__global__ void __launch_bounds__(256)
ln_res_kernel(const float* __restrict__ x, const float* __restrict__ y,
              const float* __restrict__ g, const float* __restrict__ bb,
              float* __restrict__ out) {
    __shared__ float ssum[256], ssq[256];
    const int row = (int)blockIdx.x;
    const int tid = (int)threadIdx.x;
    const float* yr = y + (size_t)row * DMOD;
    const float a0 = yr[tid], a1 = yr[tid + 256];
    ssum[tid] = a0 + a1;
    ssq[tid]  = a0 * a0 + a1 * a1;
    __syncthreads();
    for (int off = 128; off > 0; off >>= 1) {
        if (tid < off) { ssum[tid] += ssum[tid + off]; ssq[tid] += ssq[tid + off]; }
        __syncthreads();
    }
    const float mu  = ssum[0] * (1.0f / DMOD);
    const float var = ssq[0] * (1.0f / DMOD) - mu * mu;
    const float inv = rsqrtf(var + 1e-6f);
    const size_t ro = (size_t)row * DMOD;
    out[ro + tid]       = x[ro + tid]       + (a0 - mu) * inv * g[tid]       + bb[tid];
    out[ro + tid + 256] = x[ro + tid + 256] + (a1 - mu) * inv * g[tid + 256] + bb[tid + 256];
}

extern "C" void kernel_launch(void* const* d_in, const int* in_sizes, int n_in,
                              void* d_out, int out_size, void* d_ws, size_t ws_size,
                              hipStream_t stream) {
    const float* x        = (const float*)d_in[0];
    const float* in_w     = (const float*)d_in[1];   // [2048, 512]
    const float* out_w    = (const float*)d_in[2];   // [512, 1024]
    const float* conv_wf  = (const float*)d_in[3];
    const float* conv_bf  = (const float*)d_in[4];
    const float* xproj_f  = (const float*)d_in[5];   // [64, 1024]
    const float* dtproj_f = (const float*)d_in[6];   // [1024, 32]
    const float* dtbias_f = (const float*)d_in[7];
    const float* Alog_f   = (const float*)d_in[8];
    const float* D_f      = (const float*)d_in[9];
    const float* conv_wb  = (const float*)d_in[10];
    const float* conv_bb  = (const float*)d_in[11];
    const float* xproj_b  = (const float*)d_in[12];
    const float* dtproj_b = (const float*)d_in[13];
    const float* dtbias_b = (const float*)d_in[14];
    const float* Alog_b   = (const float*)d_in[15];
    const float* D_b      = (const float*)d_in[16];
    const float* ln_g     = (const float*)d_in[17];
    const float* ln_b     = (const float*)d_in[18];
    float* out = (float*)d_out;

    // ---- workspace layout: fp32 region then bf16 (ushort) region ----
    float* ws   = (float*)d_ws;
    size_t off  = 0;
    float* xz   = ws + off; off += (size_t)ROWS * 2048;
    float* xc32f= ws + off; off += (size_t)ROWS * DINR;
    float* xc32b= ws + off; off += (size_t)ROWS * DINR;
    float* dblf = ws + off; off += (size_t)ROWS * 64;
    float* dblb = ws + off; off += (size_t)ROWS * 64;
    float* dtf  = ws + off; off += (size_t)ROWS * DINR;
    float* dtb  = ws + off; off += (size_t)ROWS * DINR;
    float* u32  = ws + off; off += (size_t)ROWS * DINR;
    float* y2   = ws + off; off += (size_t)ROWS * DMOD;

    unsigned short* bws = (unsigned short*)(ws + off);
    size_t boff = 0;
    unsigned short* xbf      = bws + boff; boff += (size_t)ROWS * DMOD;
    unsigned short* inw_bf   = bws + boff; boff += (size_t)2048 * DMOD;
    unsigned short* outw_bf  = bws + boff; boff += (size_t)DMOD * DINR;
    unsigned short* xprojf_bf= bws + boff; boff += (size_t)64 * DINR;
    unsigned short* xprojb_bf= bws + boff; boff += (size_t)64 * DINR;
    unsigned short* dtprjf_bf= bws + boff; boff += (size_t)DINR * DTRK;
    unsigned short* dtprjb_bf= bws + boff; boff += (size_t)DINR * DTRK;
    unsigned short* xcf_bf   = bws + boff; boff += (size_t)ROWS * DINR;
    unsigned short* xcb_bf   = bws + boff; boff += (size_t)ROWS * DINR;
    unsigned short* dblf_bf  = bws + boff; boff += (size_t)ROWS * 64;
    unsigned short* dblb_bf  = bws + boff; boff += (size_t)ROWS * 64;
    unsigned short* u_bf     = bws + boff; boff += (size_t)ROWS * DINR;

    const dim3 blk(256);
    #define CVT(src, dst, n) \
        cvt_bf16_kernel<<<dim3((unsigned)(((n) + 255) / 256)), blk, 0, stream>>>(src, dst, (int)(n))
    // strips of 16x64; 8 waves per block
    #define GEMM_GRID(M, N) dim3((unsigned)(((M) / 16) * ((N) / 64) / 8))

    // 0) stage GEMM operands in bf16 (one-time conversions)
    CVT(x,        xbf,       (size_t)ROWS * DMOD);
    CVT(in_w,     inw_bf,    (size_t)2048 * DMOD);
    CVT(out_w,    outw_bf,   (size_t)DMOD * DINR);
    CVT(xproj_f,  xprojf_bf, (size_t)64 * DINR);
    CVT(xproj_b,  xprojb_bf, (size_t)64 * DINR);
    CVT(dtproj_f, dtprjf_bf, (size_t)DINR * DTRK);
    CVT(dtproj_b, dtprjb_bf, (size_t)DINR * DTRK);

    // 1) in_proj: xz = x @ in_w^T   [8192,2048] fp32
    gemm_wmma_kernel<<<GEMM_GRID(ROWS, 2048), blk, 0, stream>>>(
        xbf, DMOD, inw_bf, DMOD, xz, 2048, ROWS, 2048, DMOD, 0, nullptr, nullptr);

    // 2) causal conv + SiLU, both directions (fp32 + bf16 outputs)
    const dim3 convGrid((unsigned)((ROWS * DINR) / 256));
    conv_silu_kernel<<<convGrid, blk, 0, stream>>>(xz, conv_wf, conv_bf, xc32f, xcf_bf, 0);
    conv_silu_kernel<<<convGrid, blk, 0, stream>>>(xz, conv_wb, conv_bb, xc32b, xcb_bf, 1);

    // 3) x_proj: dbl = xc @ xproj^T  [8192,64]; fp32 (scan) + bf16 (dt GEMM)
    gemm_wmma_kernel<<<GEMM_GRID(ROWS, 64), blk, 0, stream>>>(
        xcf_bf, DINR, xprojf_bf, DINR, dblf, 64, ROWS, 64, DINR, 2, nullptr, dblf_bf);
    gemm_wmma_kernel<<<GEMM_GRID(ROWS, 64), blk, 0, stream>>>(
        xcb_bf, DINR, xprojb_bf, DINR, dblb, 64, ROWS, 64, DINR, 2, nullptr, dblb_bf);

    // 4) dt_proj + bias + softplus: dt = softplus(dbl[:, :32] @ dtproj^T + b)
    gemm_wmma_kernel<<<GEMM_GRID(ROWS, DINR), blk, 0, stream>>>(
        dblf_bf, 64, dtprjf_bf, DTRK, dtf, DINR, ROWS, DINR, DTRK, 1, dtbias_f, nullptr);
    gemm_wmma_kernel<<<GEMM_GRID(ROWS, DINR), blk, 0, stream>>>(
        dblb_bf, 64, dtprjb_bf, DTRK, dtb, DINR, ROWS, DINR, DTRK, 1, dtbias_b, nullptr);

    // 5) selective scans: forward writes u32, backward adds (time-flipped)
    //    and emits final bf16 u for out_proj
    const dim3 scanGrid((unsigned)(NB * (DINR / 16)));
    scan_kernel<<<scanGrid, blk, 0, stream>>>(dtf, xc32f, dblf, xz, Alog_f, D_f, u32, u_bf, 0);
    scan_kernel<<<scanGrid, blk, 0, stream>>>(dtb, xc32b, dblb, xz, Alog_b, D_b, u32, u_bf, 1);

    // 6) out_proj: y2 = u @ out_w^T   [8192,512]
    gemm_wmma_kernel<<<GEMM_GRID(ROWS, DMOD), blk, 0, stream>>>(
        u_bf, DINR, outw_bf, DINR, y2, DMOD, ROWS, DMOD, DINR, 0, nullptr, nullptr);

    // 7) residual + layernorm
    ln_res_kernel<<<dim3((unsigned)ROWS), blk, 0, stream>>>(x, y2, ln_g, ln_b, out);

    #undef CVT
    #undef GEMM_GRID
}
